// MultiHeadAttention_5016521801873
// MI455X (gfx1250) — compile-verified, matrix operand order checked by reading, not run on hardware
//
#include <hip/hip_runtime.h>
#include <hip/hip_bf16.h>
#include <stdint.h>

// ---------------------------------------------------------------------------
// MultiHeadAttention (no softmax) for gfx1250 / MI455X.
// (Q K^T / sqrt(d)) V == Q (K^T V) / sqrt(d): attention collapses to a
// per-head 64x64 matrix; everything else is dense GEMM on
// v_wmma_f32_16x16x32_bf16 (fp32 accumulate).
// LDS tiles hold bf16 pairs packed in u32 (matches the A-fragment K-pair
// packing); all global staging is float4 (global_load_b128).
// ---------------------------------------------------------------------------

typedef __attribute__((ext_vector_type(16))) __bf16 v16bf;
typedef __attribute__((ext_vector_type(8)))  float  v8f;

static constexpr int Bsz  = 4;
static constexpr int S    = 2048;
static constexpr int Dm   = 1024;   // model dim == GEMM N == GEMM K
static constexpr int Hh   = 16;
static constexpr int DK   = 64;
static constexpr int ROWS = Bsz * S;  // 8192

__device__ __forceinline__ uint32_t f2bf_u(float x) {
  union { float f; uint32_t u; } c; c.f = x;
  const uint32_t u = c.u;
  return (u + 0x7FFFu + ((u >> 16) & 1u)) >> 16;  // RNE fp32 -> bf16
}
__device__ __forceinline__ uint32_t pack2(float lo, float hi) {
  return f2bf_u(lo) | (f2bf_u(hi) << 16);  // K-even in [15:0], K-odd in [31:16]
}

// 16x32 bf16 A-fragment (also used for B^T, same layout) from packed-pair LDS.
// ISA 7.12.2: lane half h, VGPR i holds K-pair kp = (i%4) + (i/4)*8 + h*4.
union FragU { v16bf v; uint32_t u[8]; };

__device__ __forceinline__ v16bf load_frag(const uint32_t* base, int ldp, int lane) {
  FragU f;
  const int h = lane >> 4;
  const int m = lane & 15;
#pragma unroll
  for (int i = 0; i < 8; ++i) {
    const int kp = (i & 3) + ((i >> 2) << 3) + (h << 2);
    f.u[i] = base[m * ldp + kp];
  }
  return f.v;
}

// ---------------------------------------------------------------------------
// C[M,1024] = A[M,1024] @ B[1024,1024] + bias. Tile 128x128, K-step 32.
// 256 threads = 8 waves (4x2); each wave a 32x64 sub-tile (2x4 accumulators).
// ---------------------------------------------------------------------------
#define BM 128
#define BN 128
#define BK 32
#define KPAIR (BK / 2)     // 16 u32 pairs per row
#define LDP (KPAIR + 1)    // 17 (odd -> bank friendly)

__global__ __launch_bounds__(256)
void gemm_bias_kernel(const float* __restrict__ A, const float* __restrict__ B,
                      const float* __restrict__ bias, float* __restrict__ C) {
  __shared__ uint32_t Asp[BM * LDP];   // A tile  [m][kp]
  __shared__ uint32_t Btsp[BN * LDP];  // B^T tile [n][kp]

  const int tid  = threadIdx.x;
  const int wave = tid >> 5;
  const int lane = tid & 31;
  const int row0 = blockIdx.x * BM;
  const int col0 = blockIdx.y * BN;
  const int wm   = (wave >> 1) * 32;
  const int wn   = (wave & 1) * 64;

  v8f acc[2][4] = {};

  for (int k0 = 0; k0 < Dm; k0 += BK) {
    // Stage A: 128x32 floats as float4 loads, pack K-pairs.
#pragma unroll
    for (int r = 0; r < 4; ++r) {
      const int idx = tid + r * 256;        // 0..1023 float4s
      const int ar  = idx >> 3;             // 8 float4 per row
      const int c4  = (idx & 7) << 2;       // float offset in row
      const float4 f = *reinterpret_cast<const float4*>(
          &A[(size_t)(row0 + ar) * Dm + k0 + c4]);
      Asp[ar * LDP + (c4 >> 1)]     = pack2(f.x, f.y);
      Asp[ar * LDP + (c4 >> 1) + 1] = pack2(f.z, f.w);
    }
    // Stage B transposed: per unit, two float4 rows (k even/odd) at 4
    // consecutive columns -> 4 packed pairs. 512 units, 2 per thread.
#pragma unroll
    for (int r = 0; r < 2; ++r) {
      const int u  = tid + r * 256;         // 0..511
      const int n4 = (u & 31) << 2;         // 32 column-quads
      const int kp = u >> 5;                // 0..15
      const float* bp = &B[(size_t)(k0 + 2 * kp) * Dm + col0 + n4];
      const float4 f0 = *reinterpret_cast<const float4*>(bp);
      const float4 f1 = *reinterpret_cast<const float4*>(bp + Dm);
      Btsp[(n4 + 0) * LDP + kp] = pack2(f0.x, f1.x);
      Btsp[(n4 + 1) * LDP + kp] = pack2(f0.y, f1.y);
      Btsp[(n4 + 2) * LDP + kp] = pack2(f0.z, f1.z);
      Btsp[(n4 + 3) * LDP + kp] = pack2(f0.w, f1.w);
    }
    // Prefetch next K-tile (global_prefetch_b8).
    if (k0 + BK < Dm) {
      __builtin_prefetch(&A[(size_t)(row0 + (tid >> 1)) * Dm + k0 + BK], 0, 1);
      __builtin_prefetch(&B[(size_t)(k0 + BK + (tid >> 7)) * Dm + col0 + (tid & 127)], 0, 1);
    }
    __syncthreads();

    v16bf afr[2], bfr[4];
#pragma unroll
    for (int i = 0; i < 2; ++i) afr[i] = load_frag(&Asp[(wm + i * 16) * LDP], LDP, lane);
#pragma unroll
    for (int j = 0; j < 4; ++j) bfr[j] = load_frag(&Btsp[(wn + j * 16) * LDP], LDP, lane);
#pragma unroll
    for (int i = 0; i < 2; ++i)
#pragma unroll
      for (int j = 0; j < 4; ++j)
        acc[i][j] = __builtin_amdgcn_wmma_f32_16x16x32_bf16(
            false, afr[i], false, bfr[j], (short)0, acc[i][j], false, false);
    __syncthreads();
  }

  // Epilogue: hoisted bias, unconditional stores.
  const int h = lane >> 4, n = lane & 15;
  float bvals[4];
#pragma unroll
  for (int j = 0; j < 4; ++j) bvals[j] = bias[col0 + wn + j * 16 + n];

#pragma unroll
  for (int i = 0; i < 2; ++i)
#pragma unroll
    for (int j = 0; j < 4; ++j)
#pragma unroll
      for (int r = 0; r < 8; ++r) {
        const int row = row0 + wm + i * 16 + r + h * 8;
        const int col = col0 + wn + j * 16 + n;
        C[(size_t)row * Dm + col] = acc[i][j][r] + bvals[j];
      }
}

// ---------------------------------------------------------------------------
// Per (b,h): Mh[64,64] = (K_h^T @ V_h) * 0.125. One block / head, 512 threads
// = 16 waves, one 16x16 tile each, streaming S in chunks of 32.
// ---------------------------------------------------------------------------
__global__ __launch_bounds__(512)
void kmv_kernel(const float* __restrict__ Kp, const float* __restrict__ Vp,
                float* __restrict__ Mh) {
  const int bh = blockIdx.x;
  const int b = bh >> 4, hh = bh & 15;
  const float* Kh = Kp + (size_t)b * S * Dm + hh * DK;
  const float* Vh = Vp + (size_t)b * S * Dm + hh * DK;

  __shared__ uint32_t Ktp[64 * LDP];  // A = K^T: [m][kp], pair over sequence
  __shared__ uint32_t Vtp[64 * LDP];  // B^T:     [n][kp]

  const int tid  = threadIdx.x;
  const int wave = tid >> 5, lane = tid & 31;
  const int wm = (wave >> 2) * 16;
  const int wn = (wave & 3) * 16;

  v8f acc = {};
  for (int s0 = 0; s0 < S; s0 += BK) {
    // 1024 pairs each for K and V = 256 float4-pair units each.
    // Threads 0..255 stage K, 256..511 stage V: 1 unit per thread.
    {
      const int u  = tid & 255;
      const int m4 = (u & 15) << 2;       // 16 column-quads over d_k=64
      const int kp = u >> 4;              // 0..15
      const float* src = (tid < 256) ? Kh : Vh;
      uint32_t* dst    = (tid < 256) ? Ktp : Vtp;
      const float* p = &src[(size_t)(s0 + 2 * kp) * Dm + m4];
      const float4 f0 = *reinterpret_cast<const float4*>(p);
      const float4 f1 = *reinterpret_cast<const float4*>(p + Dm);
      dst[(m4 + 0) * LDP + kp] = pack2(f0.x, f1.x);
      dst[(m4 + 1) * LDP + kp] = pack2(f0.y, f1.y);
      dst[(m4 + 2) * LDP + kp] = pack2(f0.z, f1.z);
      dst[(m4 + 3) * LDP + kp] = pack2(f0.w, f1.w);
    }
    __syncthreads();
    const v16bf afr = load_frag(&Ktp[wm * LDP], LDP, lane);
    const v16bf bfr = load_frag(&Vtp[wn * LDP], LDP, lane);
    acc = __builtin_amdgcn_wmma_f32_16x16x32_bf16(
        false, afr, false, bfr, (short)0, acc, false, false);
    __syncthreads();
  }

  const int h2 = lane >> 4, n = lane & 15;
  float* Mout = Mh + (size_t)bh * DK * DK;
#pragma unroll
  for (int r = 0; r < 8; ++r)
    Mout[(wm + r + h2 * 8) * DK + (wn + n)] = acc[r] * 0.125f;  // 1/sqrt(64)
}

// ---------------------------------------------------------------------------
// Per (b,h, 128-row stripe): O_h = Q_h[128,64] @ Mh[64,64], written into
// column slot h*64 of [ROWS, 1024] (head concat for free). 8 waves, wave w
// owns m-subtile w x 4 n-tiles.
// ---------------------------------------------------------------------------
#define LDQ 33  // 32 pairs + 1 pad
__global__ __launch_bounds__(256)
void qm_kernel(const float* __restrict__ Qp, const float* __restrict__ Mh,
               float* __restrict__ O) {
  const int bh = blockIdx.x, st = blockIdx.y;
  const int b = bh >> 4, hh = bh & 15;
  const float* Qh = Qp + (size_t)b * S * Dm + hh * DK;
  const float* Mi = Mh + (size_t)bh * DK * DK;
  float* Oh = O + (size_t)b * S * Dm + hh * DK;

  __shared__ uint32_t Qsp[128 * LDQ];  // [m][kp]
  __shared__ uint32_t Mtp[64 * LDQ];   // M^T: [n][kp]

  const int tid = threadIdx.x;
  const int wave = tid >> 5, lane = tid & 31;
  const int row0 = st * 128;

  // Stage Q: 128x64 floats as float4 -> 2048 loads, 8 per thread.
#pragma unroll
  for (int r = 0; r < 8; ++r) {
    const int idx = tid + r * 256;
    const int qr = idx >> 4;            // 16 float4 per 64-wide row
    const int c4 = (idx & 15) << 2;
    const float4 f = *reinterpret_cast<const float4*>(
        &Qh[(size_t)(row0 + qr) * Dm + c4]);
    Qsp[qr * LDQ + (c4 >> 1)]     = pack2(f.x, f.y);
    Qsp[qr * LDQ + (c4 >> 1) + 1] = pack2(f.z, f.w);
  }
  // Stage M transposed: 2048 pairs = 512 float4-pair units, 2 per thread.
#pragma unroll
  for (int r = 0; r < 2; ++r) {
    const int u  = tid + r * 256;       // 0..511
    const int n4 = (u & 15) << 2;       // 16 column-quads over 64
    const int kp = u >> 4;              // 0..31
    const float* p = &Mi[(size_t)(2 * kp) * DK + n4];
    const float4 f0 = *reinterpret_cast<const float4*>(p);
    const float4 f1 = *reinterpret_cast<const float4*>(p + DK);
    Mtp[(n4 + 0) * LDQ + kp] = pack2(f0.x, f1.x);
    Mtp[(n4 + 1) * LDQ + kp] = pack2(f0.y, f1.y);
    Mtp[(n4 + 2) * LDQ + kp] = pack2(f0.z, f1.z);
    Mtp[(n4 + 3) * LDQ + kp] = pack2(f0.w, f1.w);
  }
  __syncthreads();

  const int wm = wave * 16;
  v8f acc[4] = {};
#pragma unroll
  for (int k0 = 0; k0 < DK; k0 += BK) {
    const v16bf afr = load_frag(&Qsp[wm * LDQ + (k0 >> 1)], LDQ, lane);
#pragma unroll
    for (int j = 0; j < 4; ++j) {
      const v16bf bfr = load_frag(&Mtp[(j * 16) * LDQ + (k0 >> 1)], LDQ, lane);
      acc[j] = __builtin_amdgcn_wmma_f32_16x16x32_bf16(
          false, afr, false, bfr, (short)0, acc[j], false, false);
    }
  }

  const int h2 = lane >> 4, n = lane & 15;
#pragma unroll
  for (int j = 0; j < 4; ++j)
#pragma unroll
    for (int r = 0; r < 8; ++r)
      Oh[(size_t)(row0 + wm + r + h2 * 8) * Dm + (j * 16 + n)] = acc[j][r];
}

// ---------------------------------------------------------------------------
extern "C" void kernel_launch(void* const* d_in, const int* in_sizes, int n_in,
                              void* d_out, int out_size, void* d_ws, size_t ws_size,
                              hipStream_t stream) {
  (void)in_sizes; (void)n_in; (void)out_size; (void)ws_size;

  const float* q  = (const float*)d_in[0];
  const float* k  = (const float*)d_in[1];
  const float* v  = (const float*)d_in[2];
  const float* Wq = (const float*)d_in[3];
  const float* bq = (const float*)d_in[4];
  const float* Wk = (const float*)d_in[5];
  const float* bk = (const float*)d_in[6];
  const float* Wv = (const float*)d_in[7];
  const float* bv = (const float*)d_in[8];
  const float* Wo = (const float*)d_in[9];
  const float* bo = (const float*)d_in[10];

  const size_t MS = (size_t)ROWS * Dm;  // 8M elements
  float* ws  = (float*)d_ws;
  float* Qp  = ws;            // [8192,1024] fp32
  float* Kp  = ws + MS;
  float* Vp  = ws + 2 * MS;
  float* Obf = ws + 3 * MS;   // head-concat attention output
  float* Mhb = ws + 4 * MS;   // [64][64x64] per-(b,h) K^T V

  dim3 gproj(ROWS / BM, Dm / BN);  // 64 x 8

  gemm_bias_kernel<<<gproj, 256, 0, stream>>>(q, Wq, bq, Qp);
  gemm_bias_kernel<<<gproj, 256, 0, stream>>>(k, Wk, bk, Kp);
  gemm_bias_kernel<<<gproj, 256, 0, stream>>>(v, Wv, bv, Vp);

  kmv_kernel<<<Bsz * Hh, 512, 0, stream>>>(Kp, Vp, Mhb);

  dim3 gqm(Bsz * Hh, S / 128);  // 64 x 16
  qm_kernel<<<gqm, 256, 0, stream>>>(Qp, Mhb, Obf);

  gemm_bias_kernel<<<gproj, 256, 0, stream>>>(Obf, Wo, bo, (float*)d_out);
}
